// CorefCAModel_29549374996730
// MI455X (gfx1250) — matrix operand order repaired and checked by reading, not compiled, not run on hardware
//
#include <hip/hip_runtime.h>
#include <stdint.h>

#define B_  2
#define S_  160
#define H_  768
#define NC_ 5
#define NEG_ (-1024.0f)

typedef __attribute__((ext_vector_type(16))) __bf16 v16bf;
typedef __attribute__((ext_vector_type(8)))  float  v8f;

__device__ __forceinline__ unsigned short f2bf(float f) {
  unsigned int u = __float_as_uint(f);
  u += 0x7FFFu + ((u >> 16) & 1u);     // round-to-nearest-even
  return (unsigned short)(u >> 16);
}

// ---------------- fp32 -> bf16 convert (grid-stride) ----------------
__global__ void k_cvt_bf16(const float* __restrict__ x,
                           unsigned short* __restrict__ y, int n) {
  int i = blockIdx.x * blockDim.x + threadIdx.x;
  int stride = gridDim.x * blockDim.x;
  for (; i < n; i += stride) y[i] = f2bf(x[i]);
}

// ---------------- WMMA GEMM: C[M,N] = A[M,K] @ W[K,N] (+bias) --------
// bf16 inputs, f32 accumulate. One wave computes a 16x64 strip (4 tiles
// along N) so the A fragment is reused 4x per K-step.
// A fragment (16x32, MxK): lane<16 -> M=lane, K pairs {0..7,16..23};
//                          lane>=16 -> M=lane-16, K pairs {8..15,24..31}.
// B fragment (32x16, KxN): N = lane%16, element s -> K (s<8 ? s : s+8) + hi8.
// C/D (16x16 f32): VGPR r -> row r (lanes 0-15) / row r+8 (lanes 16-31).
__global__ void k_gemm_wmma(const unsigned short* __restrict__ A,
                            const unsigned short* __restrict__ W,
                            const float* __restrict__ bias,
                            float* __restrict__ C, int M, int K, int N) {
  int stripsN = N >> 6;                           // strips of 64 columns
  int totalWaves = (M >> 4) * stripsN;
  int wave = blockIdx.x * (blockDim.x >> 5) + (threadIdx.x >> 5);
  if (wave >= totalWaves) return;                 // wave-uniform: EXEC stays all-1s
  int tm = wave / stripsN, tn = wave % stripsN;
  int lane   = threadIdx.x & 31;
  int laneLo = lane & 15;
  int hi8    = (lane >= 16) ? 8 : 0;

  v8f acc0 = {0.f,0.f,0.f,0.f,0.f,0.f,0.f,0.f};
  v8f acc1 = acc0, acc2 = acc0, acc3 = acc0;

  // A: strength-reduced pointer; +hi8 folds the lane-half K offset.
  const unsigned short* arow = A + (size_t)(tm * 16 + laneLo) * K + hi8;
  // B: column n = laneLo, K offset hi8 (table order), advances by 32 rows per K-step.
  const unsigned short* brow = W + (size_t)hi8 * N + tn * 64 + laneLo;
  const size_t bstep = (size_t)32 * N;

  union F { v16bf v; uint4 q[2]; };

  for (int k0 = 0; k0 < K; k0 += 32) {
    F af;
    af.q[0] = *(const uint4*)(arow);        // K = kb+0..7   (16 B)
    af.q[1] = *(const uint4*)(arow + 16);   // K = kb+16..23 (16 B)

    F b0, b1, b2, b3;
    F* bt[4] = {&b0, &b1, &b2, &b3};        // element s -> K row (s<8 ? s : s+8) + hi8
#pragma unroll
    for (int t = 0; t < 4; ++t)
#pragma unroll
      for (int s = 0; s < 16; ++s)
        ((unsigned short*)bt[t])[s] = brow[(size_t)(s < 8 ? s : s + 8) * N + t * 16];

    __builtin_prefetch(brow + bstep, 0, 0); // speculative: safe past end

    acc0 = __builtin_amdgcn_wmma_f32_16x16x32_bf16(false, af.v, false, b0.v,
                                                   (short)0, acc0, false, false);
    acc1 = __builtin_amdgcn_wmma_f32_16x16x32_bf16(false, af.v, false, b1.v,
                                                   (short)0, acc1, false, false);
    acc2 = __builtin_amdgcn_wmma_f32_16x16x32_bf16(false, af.v, false, b2.v,
                                                   (short)0, acc2, false, false);
    acc3 = __builtin_amdgcn_wmma_f32_16x16x32_bf16(false, af.v, false, b3.v,
                                                   (short)0, acc3, false, false);

    arow += 32;
    brow += bstep;
  }

  v8f accs[4] = {acc0, acc1, acc2, acc3};
#pragma unroll
  for (int t = 0; t < 4; ++t) {
    int n = tn * 64 + t * 16 + laneLo;
    float bv = bias ? bias[n] : 0.0f;
#pragma unroll
    for (int r = 0; r < 8; ++r) {
      int m = tm * 16 + r + hi8;
      C[(size_t)m * N + n] = accs[t][r] + bv;
    }
  }
}

// ---------------- coref logits: out[b,i,j] = W_out . tanh(hp_c[b,i]+ha_c[b,j]) + maskNEG
__global__ void k_coref(const float* __restrict__ h_p, const float* __restrict__ h_a,
                        const float* __restrict__ Wout,
                        const unsigned char* __restrict__ ng, const int* __restrict__ att,
                        float* __restrict__ outc) {
  __shared__ float hp_s[H_];
  int bi = blockIdx.x;                 // b*S + i
  int b  = bi / S_;
  const float* hp = h_p + ((size_t)bi * NC_ + (NC_ - 1)) * H_;
  for (int t = threadIdx.x; t < H_; t += blockDim.x) hp_s[t] = hp[t];
  __syncthreads();
  int wave = threadIdx.x >> 5, lane = threadIdx.x & 31;
  int nwaves = blockDim.x >> 5;
  for (int j = wave; j < S_; j += nwaves) {
    const float* ha = h_a + ((size_t)(b * S_ + j) * NC_ + (NC_ - 1)) * H_;
    float acc = 0.f;
    for (int h = lane; h < H_; h += 32)
      acc += tanhf(hp_s[h] + ha[h]) * Wout[h];
#pragma unroll
    for (int off = 16; off > 0; off >>= 1) acc += __shfl_xor(acc, off, 32);
    if (lane == 0) {
      bool m = (ng[((size_t)bi * NC_ + (NC_ - 1)) * S_ + j] != 0) && (att[b * S_ + j] > 0);
      outc[(size_t)bi * S_ + j] = acc + (m ? 0.f : NEG_);
    }
  }
}

// ---------------- row softmax (over j), emits bf16 weights ----------------
__global__ void k_softmax(const float* __restrict__ x, unsigned short* __restrict__ y) {
  __shared__ float red[256];
  int row = blockIdx.x, t = threadIdx.x;
  float v = (t < S_) ? x[(size_t)row * S_ + t] : -3.0e38f;
  red[t] = v; __syncthreads();
  for (int s = 128; s > 0; s >>= 1) { if (t < s) red[t] = fmaxf(red[t], red[t + s]); __syncthreads(); }
  float mx = red[0]; __syncthreads();
  float e = (t < S_) ? __expf(v - mx) : 0.f;
  red[t] = e; __syncthreads();
  for (int s = 128; s > 0; s >>= 1) { if (t < s) red[t] += red[t + s]; __syncthreads(); }
  float inv = 1.0f / red[0];
  if (t < S_) y[(size_t)row * S_ + t] = f2bf(e * inv);
}

// ---------------- h_a[b,j,c,:] += context[b,j,:] (in place) ----------------
__global__ void k_addctx(float* __restrict__ ha, const float* __restrict__ ctx, int n) {
  int i = blockIdx.x * blockDim.x + threadIdx.x;
  int stride = gridDim.x * blockDim.x;
  for (; i < n; i += stride) {
    int h  = i % H_;
    int bs = i / (H_ * NC_);
    ha[i] += ctx[(size_t)bs * H_ + h];
  }
}

// ---------------- main logits: out[b,i,c,j] = W_out . tanh(hp[b,i,c]+ha'[b,j,c]) + maskNEG
__global__ void k_main(const float* __restrict__ h_p, const float* __restrict__ h_a,
                       const float* __restrict__ Wout,
                       const unsigned char* __restrict__ ng, const int* __restrict__ att,
                       float* __restrict__ out) {
  __shared__ float hp_s[H_];
  int bic = blockIdx.x;                // (b*S+i)*NC + c
  int c  = bic % NC_;
  int bi = bic / NC_;
  int b  = bi / S_;
  const float* hp = h_p + (size_t)bic * H_;
  for (int t = threadIdx.x; t < H_; t += blockDim.x) hp_s[t] = hp[t];
  __syncthreads();
  int wave = threadIdx.x >> 5, lane = threadIdx.x & 31;
  int nwaves = blockDim.x >> 5;
  for (int j = wave; j < S_; j += nwaves) {
    const float* ha = h_a + ((size_t)(b * S_ + j) * NC_ + c) * H_;
    float acc = 0.f;
    for (int h = lane; h < H_; h += 32)
      acc += tanhf(hp_s[h] + ha[h]) * Wout[h];
#pragma unroll
    for (int off = 16; off > 0; off >>= 1) acc += __shfl_xor(acc, off, 32);
    if (lane == 0) {
      bool m = (ng[(size_t)bic * S_ + j] != 0) && (att[b * S_ + j] > 0);
      out[(size_t)bic * S_ + j] = acc + (m ? 0.f : NEG_);
    }
  }
}

// ---------------- loss pass 1: per-row logsumexp + target partials ----------------
__global__ void k_loss_rows(const float* __restrict__ x, const unsigned char* __restrict__ tgt,
                            float* __restrict__ part, float* __restrict__ cnt) {
  __shared__ float red[256];
  int row = blockIdx.x, t = threadIdx.x;
  float v = (t < S_) ? x[(size_t)row * S_ + t] : -3.0e38f;
  red[t] = v; __syncthreads();
  for (int s = 128; s > 0; s >>= 1) { if (t < s) red[t] = fmaxf(red[t], red[t + s]); __syncthreads(); }
  float mx = red[0]; __syncthreads();
  float e = (t < S_) ? __expf(v - mx) : 0.f;
  red[t] = e; __syncthreads();
  for (int s = 128; s > 0; s >>= 1) { if (t < s) red[t] += red[t + s]; __syncthreads(); }
  float lse = __logf(red[0]) + mx; __syncthreads();
  float tg = (t < S_ && tgt[(size_t)row * S_ + t] != 0) ? 1.f : 0.f;
  red[t] = tg * (v - lse); __syncthreads();
  for (int s = 128; s > 0; s >>= 1) { if (t < s) red[t] += red[t + s]; __syncthreads(); }
  float p = red[0]; __syncthreads();
  red[t] = tg; __syncthreads();
  for (int s = 128; s > 0; s >>= 1) { if (t < s) red[t] += red[t + s]; __syncthreads(); }
  if (t == 0) { part[row] = p; cnt[row] = red[0]; }
}

// ---------------- loss pass 2: final deterministic reduce ----------------
__global__ void k_loss_final(const float* __restrict__ part, const float* __restrict__ cnt,
                             float* __restrict__ loss, int nrows) {
  __shared__ float s1[256], s2[256];
  int t = threadIdx.x;
  float a = 0.f, b = 0.f;
  for (int i = t; i < nrows; i += 256) { a += part[i]; b += cnt[i]; }
  s1[t] = a; s2[t] = b; __syncthreads();
  for (int s = 128; s > 0; s >>= 1) {
    if (t < s) { s1[t] += s1[t + s]; s2[t] += s2[t + s]; }
    __syncthreads();
  }
  if (t == 0) loss[0] = -s1[0] / (s2[0] + 1e-6f);
}

extern "C" void kernel_launch(void* const* d_in, const int* in_sizes, int n_in,
                              void* d_out, int out_size, void* d_ws, size_t ws_size,
                              hipStream_t stream) {
  (void)in_sizes; (void)n_in; (void)out_size; (void)ws_size;
  const float*         seq  = (const float*)d_in[0];
  const int*           att  = (const int*)d_in[1];
  const unsigned char* ng   = (const unsigned char*)d_in[2];
  const unsigned char* tgt  = (const unsigned char*)d_in[3];
  const float*         Wctx = (const float*)d_in[4];
  const float*         bctx = (const float*)d_in[5];
  const float*         Wprd = (const float*)d_in[6];
  const float*         bprd = (const float*)d_in[7];
  const float*         Warg = (const float*)d_in[8];
  const float*         barg = (const float*)d_in[9];
  const float*         Wout = (const float*)d_in[10];
  float* out = (float*)d_out;

  char* ws = (char*)d_ws;
  size_t off = 0;
  auto carve = [&](size_t bytes) -> char* {
    char* p = ws + off;
    off += (bytes + 255) & ~(size_t)255;
    return p;
  };
  unsigned short* seq_bf  = (unsigned short*)carve((size_t)B_ * S_ * H_ * 2);
  unsigned short* wprd_bf = (unsigned short*)carve((size_t)H_ * H_ * NC_ * 2);
  unsigned short* warg_bf = (unsigned short*)carve((size_t)H_ * H_ * NC_ * 2);
  unsigned short* wctx_bf = (unsigned short*)carve((size_t)H_ * H_ * 2);
  float*          h_p     = (float*)carve((size_t)B_ * S_ * NC_ * H_ * 4);
  float*          h_a     = (float*)carve((size_t)B_ * S_ * NC_ * H_ * 4);
  float*          hctx    = (float*)carve((size_t)B_ * S_ * H_ * 4);
  unsigned short* hctx_bf = (unsigned short*)carve((size_t)B_ * S_ * H_ * 2);
  float*          anneal  = (float*)carve((size_t)B_ * S_ * S_ * 4);
  unsigned short* attn_bf = (unsigned short*)carve((size_t)B_ * S_ * S_ * 2);
  float*          context = (float*)carve((size_t)B_ * S_ * H_ * 4);
  float*          part    = (float*)carve((size_t)B_ * S_ * NC_ * 4);
  float*          cnt     = (float*)carve((size_t)B_ * S_ * NC_ * 4);

  auto cvt = [&](const float* x, unsigned short* y, int n) {
    int blocks = (n + 255) / 256; if (blocks > 2048) blocks = 2048;
    k_cvt_bf16<<<dim3(blocks), dim3(256), 0, stream>>>(x, y, n);
  };
  auto gemm = [&](const unsigned short* A, const unsigned short* W, const float* bias,
                  float* C, int M, int K, int N) {
    int waves = (M / 16) * (N / 64);               // 16x64 strip per wave
    int blocks = (waves + 3) / 4;                  // 4 waves (128 thr) per block
    k_gemm_wmma<<<dim3(blocks), dim3(128), 0, stream>>>(A, W, bias, C, M, K, N);
  };

  // 1) bf16 conversions
  cvt(seq,  seq_bf,  B_ * S_ * H_);
  cvt(Wprd, wprd_bf, H_ * H_ * NC_);
  cvt(Warg, warg_bf, H_ * H_ * NC_);
  cvt(Wctx, wctx_bf, H_ * H_);

  // 2) projection GEMMs (WMMA)
  gemm(seq_bf, wprd_bf, bprd, h_p,  B_ * S_, H_, H_ * NC_);
  gemm(seq_bf, warg_bf, barg, h_a,  B_ * S_, H_, H_ * NC_);
  gemm(seq_bf, wctx_bf, bctx, hctx, B_ * S_, H_, H_);
  cvt(hctx, hctx_bf, B_ * S_ * H_);

  // 3) coref attention logits + softmax
  k_coref<<<dim3(B_ * S_), dim3(256), 0, stream>>>(h_p, h_a, Wout, ng, att, anneal);
  k_softmax<<<dim3(B_ * S_), dim3(256), 0, stream>>>(anneal, attn_bf);

  // 4) context = attn @ hid_context (per-batch WMMA GEMM), fold into h_a
  for (int b = 0; b < B_; ++b)
    gemm(attn_bf + (size_t)b * S_ * S_, hctx_bf + (size_t)b * S_ * H_, nullptr,
         context + (size_t)b * S_ * H_, S_, S_, H_);
  {
    int n = B_ * S_ * NC_ * H_;
    int blocks = (n + 255) / 256; if (blocks > 2048) blocks = 2048;
    k_addctx<<<dim3(blocks), dim3(256), 0, stream>>>(h_a, context, n);
  }

  // 5) pred-arg logits -> d_out[1..], then loss -> d_out[0]
  k_main<<<dim3(B_ * S_ * NC_), dim3(256), 0, stream>>>(h_p, h_a, Wout, ng, att, out + 1);
  k_loss_rows<<<dim3(B_ * S_ * NC_), dim3(256), 0, stream>>>(out + 1, tgt, part, cnt);
  k_loss_final<<<dim3(1), dim3(256), 0, stream>>>(part, cnt, out, B_ * S_ * NC_);
}
